// SelfAttention_61022895341665
// MI455X (gfx1250) — matrix operand order repaired and checked by reading, not compiled, not run on hardware
//
#include <hip/hip_runtime.h>
#include <cstdint>
#include <cstddef>

// ---- CDNA5 WMMA types ----
typedef __attribute__((ext_vector_type(16))) __bf16 v16bf;
typedef __attribute__((ext_vector_type(8)))  float  v8f;

union V16 {
    v16bf v;
    uint32_t u[8];
};

#define S_LEN  2048
#define EMBED  1024
#define NHEADS 16
#define HDIM   64
#define KT     64            // keys per softmax step (4 x 16-key subtiles)
#define QW     16            // q rows per subtile
#define UQ     2             // q subtiles per wave
#define NWAVES 4
#define QB     (QW * UQ * NWAVES) // 128 q rows per block
#define NBATCH 4

// 1/sqrt(1024) * log2(e): scores come out pre-scaled in base-2 domain
#define QSCALE 0.04508422002777998f
#define NEGS2  (-4.508422e18f)          // -1e20 * QSCALE (masked, base-2 domain)

__device__ __forceinline__ uint16_t f32_to_bf16(float x) {
    uint32_t u = __builtin_bit_cast(uint32_t, x);
    uint32_t r = (u + 0x7FFFu + ((u >> 16) & 1u)) >> 16;   // round-to-nearest-even
    return (uint16_t)r;
}
__device__ __forceinline__ uint32_t pack_bf16(float lo, float hi) {
    return (uint32_t)f32_to_bf16(lo) | ((uint32_t)f32_to_bf16(hi) << 16);
}

__device__ __forceinline__ float fast_exp2(float x) {
#if __has_builtin(__builtin_amdgcn_exp2f)
    return __builtin_amdgcn_exp2f(x);
#else
    return __expf(x * 0.6931471805599453f);
#endif
}
__device__ __forceinline__ float fast_rcp(float x) {
#if __has_builtin(__builtin_amdgcn_rcpf)
    return __builtin_amdgcn_rcpf(x);
#else
    return 1.0f / x;
#endif
}

// generic pointers to LDS carry the wave-relative LDS byte offset in the low 32 bits
__device__ __forceinline__ uint32_t lds_off_u32(const void* p) {
    return (uint32_t)(uintptr_t)p;
}

// CDNA5 async copy: 16B global -> LDS, tracked by ASYNCcnt (bypasses VGPRs)
__device__ __forceinline__ void async_load16(uint32_t lds_off, const void* gptr) {
    asm volatile("global_load_async_to_lds_b128 %0, %1, off"
                 :: "v"(lds_off), "v"(gptr)
                 : "memory");
}
#define WAIT_ASYNC(n) asm volatile("s_wait_asynccnt %0" :: "n"(n) : "memory")

// ---- wave32 16-lane row reductions via DPP16 (no LDS traffic) ----
template <int CTRL>
__device__ __forceinline__ float dpp_movf(float x) {
    return __builtin_bit_cast(float,
        __builtin_amdgcn_update_dpp(0, __builtin_bit_cast(int, x), CTRL, 0xF, 0xF, true));
}
__device__ __forceinline__ float row16_max(float x) {
    x = fmaxf(x, dpp_movf<0xB1>(x));    // quad_perm(1,0,3,2)  : xor 1
    x = fmaxf(x, dpp_movf<0x4E>(x));    // quad_perm(2,3,0,1)  : xor 2
    x = fmaxf(x, dpp_movf<0x124>(x));   // row_ror:4
    x = fmaxf(x, dpp_movf<0x128>(x));   // row_ror:8
    return x;
}
__device__ __forceinline__ float row16_sum(float x) {
    x += dpp_movf<0xB1>(x);
    x += dpp_movf<0x4E>(x);
    x += dpp_movf<0x124>(x);
    x += dpp_movf<0x128>(x);
    return x;
}

// -------------------------------------------------------------------------
// cvt_head: K [N,S,E] f32 -> Kh [N*H, S, 64] bf16 (head-major, 8 elems/thread)
// -------------------------------------------------------------------------
__global__ __launch_bounds__(256)
void cvt_head_kernel(const float* __restrict__ src, uint16_t* __restrict__ dst) {
    size_t oi = ((size_t)blockIdx.x * 256 + threadIdx.x) * 8;
    int d  = (int)(oi & 63);
    int s  = (int)((oi >> 6) & (S_LEN - 1));
    int nh = (int)(oi >> 17);                 // S*64 = 2^17
    int n = nh >> 4, h = nh & 15;
    const float* sp = src + ((size_t)n * S_LEN + s) * EMBED + h * HDIM + d;
    float4 a = *(const float4*)sp;
    float4 b = *(const float4*)(sp + 4);
    uint4 o_;
    o_.x = pack_bf16(a.x, a.y); o_.y = pack_bf16(a.z, a.w);
    o_.z = pack_bf16(b.x, b.y); o_.w = pack_bf16(b.z, b.w);
    *(uint4*)(dst + oi) = o_;
}

// -------------------------------------------------------------------------
// cvt_tr: V [N,S,E] f32 -> Vt [N*H, 64, S] bf16 (per-head transpose via LDS)
// -------------------------------------------------------------------------
__global__ __launch_bounds__(256)
void cvt_tr_kernel(const float* __restrict__ src, uint16_t* __restrict__ dst) {
    __shared__ uint16_t tb[64][66];
    const int nh = blockIdx.x;                // 64
    const int s0 = blockIdx.y * 64;           // 32 tiles
    const int n = nh >> 4, h = nh & 15;
    #pragma unroll
    for (int it = 0; it < 16; ++it) {
        int idx = threadIdx.x + it * 256;
        int sl = idx >> 6, d = idx & 63;
        tb[sl][d] = f32_to_bf16(src[((size_t)n * S_LEN + s0 + sl) * EMBED + h * HDIM + d]);
    }
    __syncthreads();
    #pragma unroll
    for (int it = 0; it < 16; ++it) {
        int idx = threadIdx.x + it * 256;
        int d = idx >> 6, sl = idx & 63;
        dst[((size_t)nh * HDIM + d) * S_LEN + s0 + sl] = tb[sl][d];
    }
}

// -------------------------------------------------------------------------
// cvt_plain: W f32 -> bf16, 8 elems/thread
// -------------------------------------------------------------------------
__global__ __launch_bounds__(256)
void cvt_plain_kernel(const float* __restrict__ src, uint16_t* __restrict__ dst, int n8) {
    int i = blockIdx.x * 256 + threadIdx.x;
    if (i >= n8) return;
    size_t oi = (size_t)i * 8;
    float4 a = *(const float4*)(src + oi);
    float4 b = *(const float4*)(src + oi + 4);
    uint4 o_;
    o_.x = pack_bf16(a.x, a.y); o_.y = pack_bf16(a.z, a.w);
    o_.z = pack_bf16(b.x, b.y); o_.w = pack_bf16(b.z, b.w);
    *(uint4*)(dst + oi) = o_;
}

// -------------------------------------------------------------------------
// Kernel 1: flash attention, bf16 WMMA, async double-buffered K/V staging.
// Each wave: 2 subtiles x 16 q rows; 64 keys per softmax step.
// -------------------------------------------------------------------------
__global__ __launch_bounds__(128)
void attn_kernel(const uint16_t* __restrict__ Vt,   // [NH, 64, S] bf16
                 const uint16_t* __restrict__ Kh,   // [NH, S, 64] bf16
                 const float*    __restrict__ Qq,   // [N, S, E] f32
                 const int*      __restrict__ mask, // [N, S]
                 uint16_t*       __restrict__ Xout) // [N*S, E] bf16
{
    __shared__ __align__(16) uint16_t kbuf[2][KT][72];      // rows=key, cols=d
    __shared__ __align__(16) uint16_t vbuf[2][HDIM][72];    // rows=d,   cols=key
    __shared__ __align__(16) uint16_t pbuf[NWAVES][UQ][QW][72];

    const int tid  = threadIdx.x;
    const int wave = tid >> 5;
    const int lane = tid & 31;
    const int hl   = lane >> 4;
    const int l15  = lane & 15;

    const int qblocks = S_LEN / QB;  // 16
    const int blk = blockIdx.x;
    const int n   = blk / (NHEADS * qblocks);
    const int rem = blk % (NHEADS * qblocks);
    const int h   = rem / qblocks;
    const int nh  = n * NHEADS + h;
    const int q0  = (rem % qblocks) * QB + wave * (QW * UQ);

    const uint16_t* kg = Kh + (size_t)nh * S_LEN * HDIM;
    const uint16_t* vg = Vt + (size_t)nh * HDIM * S_LEN;

    auto issue_tile = [&](int par, int kb) {
        #pragma unroll
        for (int i = 0; i < 4; ++i) {               // K: 512 chunks of 16B
            int c  = tid + 128 * i;
            int kr = c >> 3, seg = c & 7;
            async_load16(lds_off_u32(&kbuf[par][kr][seg * 8]),
                         kg + (size_t)(kb + kr) * HDIM + seg * 8);
        }
        #pragma unroll
        for (int i = 0; i < 4; ++i) {               // V^T: 512 chunks of 16B
            int c = tid + 128 * i;
            int d = c >> 3, seg = c & 7;
            async_load16(lds_off_u32(&vbuf[par][d][seg * 8]),
                         vg + (size_t)d * S_LEN + kb + seg * 8);
        }
    };

    issue_tile(0, 0);

    // ---- Q subtiles -> A-layout bf16 regs, pre-scaled by 1/sqrt(E)*log2(e) ----
    V16 aq[UQ][2];
    #pragma unroll
    for (int u = 0; u < UQ; ++u) {
        const float* qbase = Qq + ((size_t)n * S_LEN + q0 + u * QW + l15) * EMBED + h * HDIM;
        #pragma unroll
        for (int c = 0; c < 2; ++c)
            #pragma unroll
            for (int p = 0; p < 8; ++p) {
                int d = 2 * p + (p >= 4 ? 8 : 0) + 8 * hl + 32 * c;  // A layout
                aq[u][c].u[p] = pack_bf16(qbase[d] * QSCALE, qbase[d + 1] * QSCALE);
            }
    }

    v8f o[UQ][4];
    float mrow[UQ][8], lrow[UQ][8];
    #pragma unroll
    for (int u = 0; u < UQ; ++u) {
        #pragma unroll
        for (int j = 0; j < 4; ++j)
            #pragma unroll
            for (int r = 0; r < 8; ++r) o[u][j][r] = 0.0f;
        #pragma unroll
        for (int r = 0; r < 8; ++r) { mrow[u][r] = -INFINITY; lrow[u][r] = 0.0f; }
    }

    const int* mbase = mask + (size_t)n * S_LEN;

    for (int it = 0; it < S_LEN / KT; ++it) {
        const int kb  = it * KT;
        const int cur = it & 1;
        const bool hasnext = (kb + KT) < S_LEN;
        if (hasnext) {
            issue_tile(cur ^ 1, kb + KT);
            WAIT_ASYNC(8);               // retire current tile, next stays in flight
        } else {
            WAIT_ASYNC(0);
        }
        __syncthreads();

        // ---- scores: 4 key subtiles, B fragments shared across q subtiles ----
        v8f s[UQ][4];
        #pragma unroll
        for (int t = 0; t < 4; ++t) {
            V16 bk[2];
            int kc = t * 16 + l15;
            #pragma unroll
            for (int c = 0; c < 2; ++c)
                #pragma unroll
                for (int p = 0; p < 8; ++p) {
                    int d = 2 * p + (p >= 4 ? 8 : 0) + 8 * hl + 32 * c;   // B layout
                    bk[c].u[p] = *(const uint32_t*)&kbuf[cur][kc][d];
                }
            #pragma unroll
            for (int u = 0; u < UQ; ++u) {
                v8f z;
                #pragma unroll
                for (int r = 0; r < 8; ++r) z[r] = 0.0f;
                z = __builtin_amdgcn_wmma_f32_16x16x32_bf16(false, aq[u][0].v, false, bk[0].v,
                                                            (short)0, z, false, false);
                z = __builtin_amdgcn_wmma_f32_16x16x32_bf16(false, aq[u][1].v, false, bk[1].v,
                                                            (short)0, z, false, false);
                s[u][t] = z;
            }
        }

        // ---- mask + online softmax in base-2 domain (DPP row reductions) ----
        int mk[4];
        #pragma unroll
        for (int t = 0; t < 4; ++t) mk[t] = mbase[kb + t * 16 + l15];

        #pragma unroll
        for (int u = 0; u < UQ; ++u) {
            float alpha[8];
            #pragma unroll
            for (int r = 0; r < 8; ++r) {
                float v0 = mk[0] ? s[u][0][r] : NEGS2;
                float v1 = mk[1] ? s[u][1][r] : NEGS2;
                float v2 = mk[2] ? s[u][2][r] : NEGS2;
                float v3 = mk[3] ? s[u][3][r] : NEGS2;
                float tm = row16_max(fmaxf(fmaxf(v0, v1), fmaxf(v2, v3)));
                float mnew = fmaxf(mrow[u][r], tm);
                alpha[r] = fast_exp2(mrow[u][r] - mnew);
                mrow[u][r] = mnew;
                float p0 = fast_exp2(v0 - mnew);
                float p1 = fast_exp2(v1 - mnew);
                float p2 = fast_exp2(v2 - mnew);
                float p3 = fast_exp2(v3 - mnew);
                s[u][0][r] = p0; s[u][1][r] = p1; s[u][2][r] = p2; s[u][3][r] = p3;
                lrow[u][r] = lrow[u][r] * alpha[r] + row16_sum((p0 + p1) + (p2 + p3));
            }
            #pragma unroll
            for (int j = 0; j < 4; ++j)
                #pragma unroll
                for (int r = 0; r < 8; ++r) o[u][j][r] *= alpha[r];

            // C-layout P (f32) -> LDS (bf16, A-layout source)
            #pragma unroll
            for (int t = 0; t < 4; ++t)
                #pragma unroll
                for (int r = 0; r < 8; ++r)
                    pbuf[wave][u][r + 8 * hl][t * 16 + l15] = f32_to_bf16(s[u][t][r]);
        }
        __syncthreads();

        V16 ap[UQ][2];
        #pragma unroll
        for (int u = 0; u < UQ; ++u)
            #pragma unroll
            for (int c = 0; c < 2; ++c)
                #pragma unroll
                for (int p = 0; p < 8; ++p) {
                    int kk = 2 * p + (p >= 4 ? 8 : 0) + 8 * hl + 32 * c;  // A layout
                    ap[u][c].u[p] = *(const uint32_t*)&pbuf[wave][u][l15][kk];
                }

        // ---- O += P (16x64) * V (64x64): V fragments shared across q subtiles ----
        #pragma unroll
        for (int j = 0; j < 4; ++j) {
            V16 bv[2];
            int d = j * 16 + l15;
            #pragma unroll
            for (int c = 0; c < 2; ++c)
                #pragma unroll
                for (int p = 0; p < 8; ++p) {
                    int kk = 2 * p + (p >= 4 ? 8 : 0) + 8 * hl + 32 * c;    // B layout (key index)
                    bv[c].u[p] = *(const uint32_t*)&vbuf[cur][d][kk];
                }
            #pragma unroll
            for (int u = 0; u < UQ; ++u) {
                o[u][j] = __builtin_amdgcn_wmma_f32_16x16x32_bf16(false, ap[u][0].v, false, bv[0].v,
                                                                  (short)0, o[u][j], false, false);
                o[u][j] = __builtin_amdgcn_wmma_f32_16x16x32_bf16(false, ap[u][1].v, false, bv[1].v,
                                                                  (short)0, o[u][j], false, false);
            }
        }
        __syncthreads();
    }

    // ---- normalize (rcp) and store bf16 attention output ----
    #pragma unroll
    for (int u = 0; u < UQ; ++u) {
        float rl[8];
        #pragma unroll
        for (int r = 0; r < 8; ++r) rl[r] = fast_rcp(lrow[u][r]);
        #pragma unroll
        for (int j = 0; j < 4; ++j)
            #pragma unroll
            for (int r = 0; r < 8; ++r) {
                int qr = q0 + u * QW + r + 8 * hl;
                int e  = h * HDIM + j * 16 + l15;
                Xout[((size_t)n * S_LEN + qr) * EMBED + e] =
                    f32_to_bf16(o[u][j][r] * rl[r]);
            }
    }
}

// -------------------------------------------------------------------------
// Kernel 2: Y = X @ W^T (M=8192, N=K=1024), bf16 WMMA, async double buffer.
// Block tile 128x64, each wave 32 rows (2 subtiles), W fragments shared.
// -------------------------------------------------------------------------
__global__ __launch_bounds__(128)
void proj_kernel(const uint16_t* __restrict__ X,   // [M, E] bf16
                 const uint16_t* __restrict__ W,   // [E, E] bf16
                 float*          __restrict__ Y) { // [M, E] f32
    __shared__ __align__(16) uint16_t xbuf[2][128][40];
    __shared__ __align__(16) uint16_t wbuf[2][64][40];

    const int tid  = threadIdx.x;
    const int wave = tid >> 5;
    const int lane = tid & 31;
    const int hl   = lane >> 4;
    const int l15  = lane & 15;
    const int mb   = blockIdx.x * 128;
    const int nb   = blockIdx.y * 64;

    auto issue = [&](int par, int kb) {
        #pragma unroll
        for (int i = 0; i < 4; ++i) {              // X: 512 chunks of 16B
            int c  = tid + 128 * i;
            int rr = c >> 2, seg = c & 3;
            async_load16(lds_off_u32(&xbuf[par][rr][seg * 8]),
                         X + (size_t)(mb + rr) * EMBED + kb + seg * 8);
        }
        #pragma unroll
        for (int i = 0; i < 2; ++i) {              // W: 256 chunks of 16B
            int c  = tid + 128 * i;
            int rr = c >> 2, seg = c & 3;
            async_load16(lds_off_u32(&wbuf[par][rr][seg * 8]),
                         W + (size_t)(nb + rr) * EMBED + kb + seg * 8);
        }
    };

    v8f acc[2][4];
    #pragma unroll
    for (int u = 0; u < 2; ++u)
        #pragma unroll
        for (int j = 0; j < 4; ++j)
            #pragma unroll
            for (int r = 0; r < 8; ++r) acc[u][j][r] = 0.0f;

    issue(0, 0);
    for (int it = 0; it < EMBED / 32; ++it) {
        const int kb  = it * 32;
        const int cur = it & 1;
        const bool hasnext = (kb + 32) < EMBED;
        if (hasnext) {
            issue(cur ^ 1, kb + 32);
            WAIT_ASYNC(6);
        } else {
            WAIT_ASYNC(0);
        }
        __syncthreads();

        V16 ax[2];
        #pragma unroll
        for (int u = 0; u < 2; ++u)
            #pragma unroll
            for (int p = 0; p < 8; ++p) {
                int kk = 2 * p + (p >= 4 ? 8 : 0) + 8 * hl;   // A layout
                ax[u].u[p] = *(const uint32_t*)&xbuf[cur][wave * 32 + u * 16 + l15][kk];
            }
        #pragma unroll
        for (int j = 0; j < 4; ++j) {
            V16 bw;
            int e = j * 16 + l15;
            #pragma unroll
            for (int p = 0; p < 8; ++p) {
                int kk = 2 * p + (p >= 4 ? 8 : 0) + 8 * hl;                 // B layout
                bw.u[p] = *(const uint32_t*)&wbuf[cur][e][kk];
            }
            #pragma unroll
            for (int u = 0; u < 2; ++u)
                acc[u][j] = __builtin_amdgcn_wmma_f32_16x16x32_bf16(false, ax[u].v, false, bw.v,
                                                                    (short)0, acc[u][j], false, false);
        }
        __syncthreads();
    }

    #pragma unroll
    for (int u = 0; u < 2; ++u)
        #pragma unroll
        for (int j = 0; j < 4; ++j)
            #pragma unroll
            for (int r = 0; r < 8; ++r)
                Y[(size_t)(mb + wave * 32 + u * 16 + r + 8 * hl) * EMBED + nb + j * 16 + l15] =
                    acc[u][j][r];
}

// -------------------------------------------------------------------------
extern "C" void kernel_launch(void* const* d_in, const int* in_sizes, int n_in,
                              void* d_out, int out_size, void* d_ws, size_t ws_size,
                              hipStream_t stream) {
    // setup_inputs order: values, keys, query, mask, W_fc
    const float* vals = (const float*)d_in[0];
    const float* keys = (const float*)d_in[1];
    const float* qry  = (const float*)d_in[2];
    const int*   msk  = (const int*)d_in[3];
    const float* wfc  = (const float*)d_in[4];
    float* out = (float*)d_out;

    // workspace layout (bf16 elements): X | Kh | Vt | W  -> 50 MB total
    const size_t NSE = (size_t)NBATCH * S_LEN * EMBED;     // 8M elements
    uint16_t* wsb = (uint16_t*)d_ws;
    uint16_t* Xbf = wsb;
    uint16_t* Khb = wsb + NSE;
    uint16_t* Vtb = wsb + 2 * NSE;
    uint16_t* Wbf = wsb + 3 * NSE;

    cvt_head_kernel<<<dim3((unsigned)(NSE / 8 / 256)), dim3(256), 0, stream>>>(keys, Khb);
    cvt_tr_kernel<<<dim3(NBATCH * NHEADS, S_LEN / 64), dim3(256), 0, stream>>>(vals, Vtb);
    cvt_plain_kernel<<<dim3((EMBED * EMBED / 8 + 255) / 256), dim3(256), 0, stream>>>(
        wfc, Wbf, EMBED * EMBED / 8);

    attn_kernel<<<dim3(NBATCH * NHEADS * (S_LEN / QB)), dim3(128), 0, stream>>>(
        Vtb, Khb, qry, msk, Xbf);

    proj_kernel<<<dim3((NBATCH * S_LEN) / 128, EMBED / 64), dim3(128), 0, stream>>>(
        Xbf, Wbf, out);
}
